// MyModel_11347303596236
// MI455X (gfx1250) — matrix operand order repaired and checked by reading, not compiled, not run on hardware
//
#include <hip/hip_runtime.h>

// ---------------------------------------------------------------------------
// Hetero-GCN (3 relations, 4 layers) + edge MLP for MI455X (gfx1250).
// Dense per-relation GEMMs in bf16 via v_wmma_f32_16x16x32_bf16 (fp32 acc),
// 128x64 block tile, 32x32 wave tile (4 WMMA/wave/K-step), double-buffered
// LDS with one barrier per K-step + global_prefetch. Message passing via
// wave-per-edge float4 gather + non-returning global fp32 atomics with the
// symmetric norms folded into the message. Edge scorer: LDS-staged fp32 MLP.
// ---------------------------------------------------------------------------

typedef __attribute__((ext_vector_type(16))) __bf16 v16bf;
typedef __attribute__((ext_vector_type(8)))  float  v8f;

#define GN   50000   // nodes
#define GR   3       // relations
#define GE   50000   // edges per relation
#define GEP  100000  // pos/neg edges

__device__ __forceinline__ unsigned short f2bf(float f) {
  union { float f; unsigned u; } v; v.f = f;
  unsigned u = v.u;
  unsigned r = (u + 0x7FFFu + ((u >> 16) & 1u)) >> 16;   // round-to-nearest-even
  return (unsigned short)r;
}

// ---------------- elementwise helpers ----------------
__global__ void k_zero(float* __restrict__ p, long long n) {
  long long t = (long long)blockIdx.x * blockDim.x + threadIdx.x;
  if (t < n) p[t] = 0.0f;
}

__global__ void k_f2bf(const float* __restrict__ in, unsigned short* __restrict__ out, long long n) {
  long long t = (long long)blockIdx.x * blockDim.x + threadIdx.x;
  if (t < n) out[t] = f2bf(in[t]);
}

__global__ void k_deg(const int* __restrict__ src, const int* __restrict__ dst,
                      float* __restrict__ od, float* __restrict__ id, int RE, int E) {
  int t = blockIdx.x * blockDim.x + threadIdx.x;
  if (t >= RE) return;
  int r = t / E;
  unsafeAtomicAdd(&od[(size_t)r * GN + src[t]], 1.0f);
  unsafeAtomicAdd(&id[(size_t)r * GN + dst[t]], 1.0f);
}

__global__ void k_norm(float* __restrict__ deg, long long n) {
  long long t = (long long)blockIdx.x * blockDim.x + threadIdx.x;
  if (t < n) deg[t] = rsqrtf(fmaxf(deg[t], 1.0f));
}

// out[n][c] = sum_r b[r][c]   (bias broadcast baseline before scatter-adds)
__global__ void k_bias_init(const float* __restrict__ b, float* __restrict__ outb,
                            long long total, int dout) {
  long long t = (long long)blockIdx.x * blockDim.x + threadIdx.x;
  if (t >= total) return;
  int c = (int)(t % dout);
  outb[t] = b[c] + b[dout + c] + b[2 * dout + c];
}

// relu (optional) + write fp32 back + bf16 copy for next layer's GEMM
__global__ void k_finalize(float* __restrict__ outb, unsigned short* __restrict__ hb,
                           long long n, int doRelu) {
  long long t = (long long)blockIdx.x * blockDim.x + threadIdx.x;
  if (t >= n) return;
  float v = outb[t];
  if (doRelu) v = fmaxf(v, 0.0f);
  outb[t] = v;
  hb[t] = f2bf(v);
}

// one wave32 per edge: msg = hr[src][:]*sn[src]*dn[dst] -> atomic add outb[dst]
__global__ __launch_bounds__(256)
void k_scatter(const float* __restrict__ hr, const int* __restrict__ src,
               const int* __restrict__ dst, const float* __restrict__ sn,
               const float* __restrict__ dn, float* __restrict__ outb,
               int E, int dout) {
  const int lane = threadIdx.x & 31;
  const int wave = threadIdx.x >> 5;
  const int d4 = dout >> 2;
  for (int e = blockIdx.x * 8 + wave; e < E; e += gridDim.x * 8) {
    int s = src[e], d = dst[e];
    float scale = sn[s] * dn[d];
    const float4* row = (const float4*)(hr + (size_t)s * dout);
    float* orow = outb + (size_t)d * dout;
    for (int c4 = lane; c4 < d4; c4 += 32) {
      float4 v = row[c4];
      int c = c4 * 4;
      unsafeAtomicAdd(&orow[c],     v.x * scale);
      unsafeAtomicAdd(&orow[c + 1], v.y * scale);
      unsafeAtomicAdd(&orow[c + 2], v.z * scale);
      unsafeAtomicAdd(&orow[c + 3], v.w * scale);
    }
  }
}

// ---------------- bf16 WMMA GEMM: C[M,Nc] = A[M,K] * B[K,Nc] ----------------
// Block tile 128x64, 8 waves as 4(M)x2(N); each wave: 32x32 = 2x2 subtiles
// -> 4 v_wmma_f32_16x16x32_bf16 per K-step. Double-buffered LDS, 1 barrier/step.
#define BM 128
#define BN 64
#define BK 32

__device__ __forceinline__ v8f wmma_bf16(const v16bf& a, const v16bf& b, const v8f& c) {
  return __builtin_amdgcn_wmma_f32_16x16x32_bf16(false, a, false, b, (short)0, c, false, false);
}

__global__ __launch_bounds__(256)
void k_gemm_bf16(const unsigned short* __restrict__ A, const unsigned short* __restrict__ B,
                 float* __restrict__ C, int M, int K, int Nc) {
  __shared__ unsigned short As[2][BM][BK];   // 2 x 8 KB
  __shared__ unsigned short Bs[2][BK][BN];   // 2 x 4 KB
  const int tid  = threadIdx.x;
  const int lane = tid & 31;
  const int wave = tid >> 5;
  const int m0 = blockIdx.x * BM;
  const int n0 = blockIdx.y * BN;
  const int wm = (wave & 3) * 32;        // wave M offset (4 waves x 32 = 128)
  const int wn = (wave >> 2) * 32;       // wave N offset (2 waves x 32 = 64)

  v8f acc00 = {}, acc01 = {}, acc10 = {}, acc11 = {};

  // cooperative tile-load mapping (16B chunks)
  const int ar = tid >> 2, ac = (tid & 3) * 8;   // A: rows ar, ar+64; 4 chunks/row
  const int br = tid >> 3, bc = (tid & 7) * 8;   // B: 32 rows x 8 chunks
  const int gm0 = m0 + ar, gm1 = m0 + ar + 64;
  const int gm0c = gm0 < M ? gm0 : M - 1;
  const int gm1c = gm1 < M ? gm1 : M - 1;

  uint4 av0, av1, bv;
  auto loadg = [&](int k0) {
    av0 = make_uint4(0u, 0u, 0u, 0u);
    av1 = make_uint4(0u, 0u, 0u, 0u);
    if (gm0 < M) av0 = *(const uint4*)(A + (size_t)gm0 * K + k0 + ac);
    if (gm1 < M) av1 = *(const uint4*)(A + (size_t)gm1 * K + k0 + ac);
    bv = *(const uint4*)(B + (size_t)(k0 + br) * Nc + n0 + bc);
  };
  auto stores = [&](int buf) {
    *(uint4*)(&As[buf][ar][ac])      = av0;
    *(uint4*)(&As[buf][ar + 64][ac]) = av1;
    *(uint4*)(&Bs[buf][br][bc])      = bv;
  };

  loadg(0);
  stores(0);
  __syncthreads();

  const int am = lane & 15;
  const int khalf = (lane >> 4) * 8;

  int buf = 0;
  for (int k0 = 0; k0 < K; k0 += BK, buf ^= 1) {
    const bool more = (k0 + BK) < K;
    if (more) {
      loadg(k0 + BK);
      if (k0 + 2 * BK < K) {   // global_prefetch of the K-tile after next
        __builtin_prefetch((const void*)(A + (size_t)gm0c * K + k0 + 2 * BK + ac), 0, 1);
        __builtin_prefetch((const void*)(A + (size_t)gm1c * K + k0 + 2 * BK + ac), 0, 1);
        __builtin_prefetch((const void*)(B + (size_t)(k0 + 2 * BK + br) * Nc + n0 + bc), 0, 1);
      }
    }

    // A fragments (ISA 16-bit 16x32 layout):
    // vgpr-pair v holds K = (v>>2)*16 + (lane>=16)*8 + (v&3)*2  (+0,+1)
    union { unsigned int ui[8]; v16bf v; } a0, a1, b0, b1;
#pragma unroll
    for (int v = 0; v < 8; ++v) {
      int kp = ((v >> 2) * 16) + khalf + ((v & 3) * 2);
      a0.ui[v] = *(const unsigned int*)(&As[buf][wm + am][kp]);
      a1.ui[v] = *(const unsigned int*)(&As[buf][wm + 16 + am][kp]);
    }
    // B fragments: lane holds column N = lane&15; same K order as A
#pragma unroll
    for (int j = 0; j < 8; ++j) {
      int kq = ((j >> 2) * 16) + khalf + ((j & 3) * 2);
      b0.ui[j] = (unsigned int)Bs[buf][kq][wn + am] | ((unsigned int)Bs[buf][kq + 1][wn + am] << 16);
      b1.ui[j] = (unsigned int)Bs[buf][kq][wn + 16 + am] | ((unsigned int)Bs[buf][kq + 1][wn + 16 + am] << 16);
    }

    acc00 = wmma_bf16(a0.v, b0.v, acc00);
    acc01 = wmma_bf16(a0.v, b1.v, acc01);
    acc10 = wmma_bf16(a1.v, b0.v, acc10);
    acc11 = wmma_bf16(a1.v, b1.v, acc11);

    if (more) {
      stores(buf ^ 1);     // prev iter's barrier already ordered readers of buf^1
      __syncthreads();
    }
  }

  // C layout: VGPR v -> M = v + 8*(lane>=16), N = lane&15
  const int crow = m0 + wm + ((lane >> 4) * 8);
  const int ccol = n0 + wn + (lane & 15);
#pragma unroll
  for (int v = 0; v < 8; ++v) {
    int r0 = crow + v, r1 = r0 + 16;
    if (r0 < M) {
      C[(size_t)r0 * Nc + ccol]      = acc00[v];
      C[(size_t)r0 * Nc + ccol + 16] = acc01[v];
    }
    if (r1 < M) {
      C[(size_t)r1 * Nc + ccol]      = acc10[v];
      C[(size_t)r1 * Nc + ccol + 16] = acc11[v];
    }
  }
}

// ---------------- edge MLP: score = relu([h_s|h_d] @ Wp1 + bp1) @ Wp2 + bp2 --
__global__ __launch_bounds__(256)
void k_edge_mlp(const float* __restrict__ h, const int* __restrict__ src,
                const int* __restrict__ dst, const float* __restrict__ Wp1,
                const float* __restrict__ bp1, const float* __restrict__ Wp2,
                const float* __restrict__ bp2, float* __restrict__ out, int EP) {
  __shared__ float Ws[128 * 64];
  __shared__ float b1s[64], w2s[64];
  __shared__ float zs[8][128];
  for (int i = threadIdx.x; i < 128 * 64; i += 256) Ws[i] = Wp1[i];
  if (threadIdx.x < 64) { b1s[threadIdx.x] = bp1[threadIdx.x]; w2s[threadIdx.x] = Wp2[threadIdx.x]; }
  __syncthreads();
  const int lane = threadIdx.x & 31;
  const int wave = threadIdx.x >> 5;
  const float b2 = bp2[0];
  for (int e = blockIdx.x * 8 + wave; e < EP; e += gridDim.x * 8) {
    int s = src[e], d = dst[e];
#pragma unroll
    for (int q = 0; q < 4; ++q) {
      int i = lane + q * 32;
      zs[wave][i] = (i < 64) ? h[(size_t)s * 64 + i] : h[(size_t)d * 64 + (i - 64)];
    }
    float a0 = b1s[lane], a1 = b1s[lane + 32];
#pragma unroll 8
    for (int i = 0; i < 128; ++i) {
      float zi = zs[wave][i];
      a0 = fmaf(zi, Ws[i * 64 + lane], a0);
      a1 = fmaf(zi, Ws[i * 64 + lane + 32], a1);
    }
    a0 = fmaxf(a0, 0.0f); a1 = fmaxf(a1, 0.0f);
    float partial = a0 * w2s[lane] + a1 * w2s[lane + 32];
    for (int off = 16; off > 0; off >>= 1) partial += __shfl_down(partial, off, 32);
    if (lane == 0) out[e] = partial + b2;
  }
}

// ---------------------------------------------------------------------------
extern "C" void kernel_launch(void* const* d_in, const int* in_sizes, int n_in,
                              void* d_out, int out_size, void* d_ws, size_t ws_size,
                              hipStream_t stream) {
  const float* x       = (const float*)d_in[0];
  const int*   rel_src = (const int*)d_in[1];
  const int*   rel_dst = (const int*)d_in[2];
  const int*   pos_src = (const int*)d_in[3];
  const int*   pos_dst = (const int*)d_in[4];
  const int*   neg_src = (const int*)d_in[5];
  const int*   neg_dst = (const int*)d_in[6];
  const float* Wl[4] = { (const float*)d_in[7],  (const float*)d_in[9],
                         (const float*)d_in[11], (const float*)d_in[13] };
  const float* bl[4] = { (const float*)d_in[8],  (const float*)d_in[10],
                         (const float*)d_in[12], (const float*)d_in[14] };
  const float* Wp1 = (const float*)d_in[15];
  const float* bp1 = (const float*)d_in[16];
  const float* Wp2 = (const float*)d_in[17];
  const float* bp2 = (const float*)d_in[18];
  float* outp = (float*)d_out;

  // workspace carve-out
  size_t off = 0;
  auto carve = [&](size_t bytes) -> void* {
    void* p = (char*)d_ws + off;
    off += (bytes + 255) & ~(size_t)255;
    return p;
  };
  unsigned short* hb   = (unsigned short*)carve((size_t)GN * 512 * 2); // bf16 activations
  float*          hr   = (float*)carve((size_t)GN * 512 * 4);          // per-relation GEMM out
  float*          outb = (float*)carve((size_t)GN * 512 * 4);          // accumulated layer out
  float*          sn   = (float*)carve((size_t)GR * GN * 4);           // src norms
  float*          dn   = (float*)carve((size_t)GR * GN * 4);           // dst norms
  unsigned short* Wb   = (unsigned short*)carve((size_t)GR * 512 * 512 * 2); // bf16 weights

  const int BLK = 256;
  auto blocks = [&](long long n) { return (unsigned)((n + BLK - 1) / BLK); };

  // degree -> norms
  k_zero<<<blocks((long long)GR * GN), BLK, 0, stream>>>(sn, (long long)GR * GN);
  k_zero<<<blocks((long long)GR * GN), BLK, 0, stream>>>(dn, (long long)GR * GN);
  k_deg<<<blocks((long long)GR * GE), BLK, 0, stream>>>(rel_src, rel_dst, sn, dn, GR * GE, GE);
  k_norm<<<blocks((long long)GR * GN), BLK, 0, stream>>>(sn, (long long)GR * GN);
  k_norm<<<blocks((long long)GR * GN), BLK, 0, stream>>>(dn, (long long)GR * GN);

  // x -> bf16
  k_f2bf<<<blocks((long long)GN * 512), BLK, 0, stream>>>(x, hb, (long long)GN * 512);

  const int dims[5] = {512, 512, 256, 128, 64};
  for (int L = 0; L < 4; ++L) {
    const int din = dims[L], dout = dims[L + 1];
    const long long wcount = (long long)GR * din * dout;
    k_f2bf<<<blocks(wcount), BLK, 0, stream>>>(Wl[L], Wb, wcount);
    const long long outn = (long long)GN * dout;
    k_bias_init<<<blocks(outn), BLK, 0, stream>>>(bl[L], outb, outn, dout);
    dim3 ggrid((GN + BM - 1) / BM, dout / BN);
    for (int r = 0; r < GR; ++r) {
      k_gemm_bf16<<<ggrid, 256, 0, stream>>>(hb, Wb + (size_t)r * din * dout, hr, GN, din, dout);
      k_scatter<<<(GE + 7) / 8, 256, 0, stream>>>(hr, rel_src + (size_t)r * GE,
                                                  rel_dst + (size_t)r * GE,
                                                  sn + (size_t)r * GN, dn + (size_t)r * GN,
                                                  outb, GE, dout);
    }
    k_finalize<<<blocks(outn), BLK, 0, stream>>>(outb, hb, outn, L < 3 ? 1 : 0);
  }

  // link predictor on final h (fp32, [GN,64] in outb)
  k_edge_mlp<<<1024, 256, 0, stream>>>(outb, pos_src, pos_dst, Wp1, bp1, Wp2, bp2, outp, GEP);
  k_edge_mlp<<<1024, 256, 0, stream>>>(outb, neg_src, neg_dst, Wp1, bp1, Wp2, bp2, outp + GEP, GEP);
}
